// HookedMambaLayerBatchSplit_74929999446530
// MI455X (gfx1250) — matrix operand order repaired and checked by reading, not compiled, not run on hardware
//
#include <hip/hip_runtime.h>
#include <math.h>

typedef __attribute__((ext_vector_type(16))) __bf16 v16bf;
typedef __attribute__((ext_vector_type(8)))  __bf16 v8bf;
typedef __attribute__((ext_vector_type(8)))  float  v8f;
typedef __attribute__((ext_vector_type(4)))  int    v4i;

#define B_  2
#define L_  1024
#define D_  1024
#define E_  2048
#define N_  16
#define DT_ 64
#define KC_ 4
#define BL_ (B_*L_)
static constexpr float EPS_ = 1e-5f;

#if defined(__has_builtin)
#if __has_builtin(__builtin_amdgcn_global_load_async_to_lds_b128) && __has_builtin(__builtin_amdgcn_s_wait_asynccnt)
#define USE_ASYNC_LDS 1
#endif
#endif

__device__ __forceinline__ float sigmoidf_(float x) { return 1.0f / (1.0f + __expf(-x)); }

__device__ __forceinline__ void split_bf16(float v, __bf16* hi, __bf16* lo) {
  __bf16 h = (__bf16)v;
  *hi = h;
  *lo = (__bf16)(v - (float)h);
}

// assemble a 16-element bf16 fragment from two 8-element (16B) LDS runs
__device__ __forceinline__ v16bf frag16(const __bf16* p0, const __bf16* p1) {
  v8bf a = *(const v8bf*)p0;
  v8bf b = *(const v8bf*)p1;
  return __builtin_shufflevector(a, b, 0,1,2,3,4,5,6,7,8,9,10,11,12,13,14,15);
}

// 16B copy global -> LDS (CDNA5 async path; ASYNCcnt tracked)
__device__ __forceinline__ void copy16(const __bf16* g, __bf16* l) {
#if USE_ASYNC_LDS
  __builtin_amdgcn_global_load_async_to_lds_b128((v4i*)g, (v4i*)l, 0, 0);
#else
  *(uint4*)l = *(const uint4*)g;
#endif
}

__device__ __forceinline__ void async_wait_all() {
#if USE_ASYNC_LDS
  __builtin_amdgcn_s_wait_asynccnt(0);
#endif
}

// ---------------- RMSNorm -> bf16 hi/lo pair ----------------
__global__ __launch_bounds__(256) void rmsnorm_kernel(const float* __restrict__ resid,
                                                      const float* __restrict__ norm_w,
                                                      __bf16* __restrict__ xnhi,
                                                      __bf16* __restrict__ xnlo)
{
  const int row = blockIdx.x;
  const int tid = threadIdx.x;
  const float* r = resid + (size_t)row * D_;
  float s = 0.f;
  #pragma unroll
  for (int i = 0; i < D_ / 256; ++i) { float v = r[tid + i * 256]; s += v * v; }
  __shared__ float red[256];
  red[tid] = s; __syncthreads();
  for (int off = 128; off > 0; off >>= 1) {
    if (tid < off) red[tid] += red[tid + off];
    __syncthreads();
  }
  const float scale = rsqrtf(red[0] / (float)D_ + EPS_);
  #pragma unroll
  for (int i = 0; i < D_ / 256; ++i) {
    int j = tid + i * 256;
    float v = r[j] * scale * norm_w[j];
    split_bf16(v, &xnhi[(size_t)row * D_ + j], &xnlo[(size_t)row * D_ + j]);
  }
}

// ---------------- generic f32 -> bf16 hi/lo ----------------
__global__ __launch_bounds__(256) void cvt_pair_kernel(const float* __restrict__ x,
                                                       __bf16* __restrict__ hi,
                                                       __bf16* __restrict__ lo, int n)
{
  int i = blockIdx.x * 256 + threadIdx.x;
  if (i < n) split_bf16(x[i], &hi[i], &lo[i]);
}

// ---------------- bf16x3 WMMA GEMM: C[M,N] = epi( A[M,K] @ W[N,K]^T ) ----------------
// Block tile 128x64, 8 waves in 4(M)x2(N); each wave owns 32x32 (4 WMMA tiles).
// epi: 0=none, 1=softplus(v+bias[n]), 2=v+Radd
__global__ __launch_bounds__(256) void gemm_bf16x3_kernel(
    const __bf16* __restrict__ Ahi, const __bf16* __restrict__ Alo,
    const __bf16* __restrict__ Whi, const __bf16* __restrict__ Wlo,
    float* __restrict__ C, const float* __restrict__ bias,
    const float* __restrict__ Radd, int M, int N, int K, int epi)
{
  __shared__ __bf16 sAhi[128][48];
  __shared__ __bf16 sAlo[128][48];
  __shared__ __bf16 sWhi[64][48];
  __shared__ __bf16 sWlo[64][48];

  const int tid  = threadIdx.x;
  const int lane = tid & 31;
  const int wv   = tid >> 5;
  const int wm   = wv & 3;        // 4 waves along M (32 rows each)
  const int wn   = wv >> 2;       // 2 waves along N (32 cols each)
  const int nl   = lane & 15;
  const int hi16 = lane >> 4;
  const int m0   = blockIdx.x * 128;
  const int n0   = blockIdx.y * 64;

  v8f acc[2][2] = {};

  for (int k0 = 0; k0 < K; k0 += 32) {
    // stage W tile: 64 rows x 4 chunks = 256 chunks (one per thread)
    {
      int r = tid >> 2, q = (tid & 3) * 8;
      int gn = n0 + r;
      if (gn < N) {
        const size_t gw = (size_t)gn * K + k0 + q;
        copy16(Whi + gw, &sWhi[r][q]);
        copy16(Wlo + gw, &sWlo[r][q]);
      } else {
        uint4 z = make_uint4(0u, 0u, 0u, 0u);
        *(uint4*)&sWhi[r][q] = z;
        *(uint4*)&sWlo[r][q] = z;
      }
    }
    // stage A tile: 128 rows x 4 chunks = 512 chunks (two per thread); rows always valid
    #pragma unroll
    for (int c = 0; c < 2; ++c) {
      int j = tid + c * 256;
      int r = j >> 2, q = (j & 3) * 8;
      const size_t ga = (size_t)(m0 + r) * K + k0 + q;
      copy16(Ahi + ga, &sAhi[r][q]);
      copy16(Alo + ga, &sAlo[r][q]);
    }
    // prefetch next k-tile of A (global_prefetch)
    if (k0 + 32 < K) __builtin_prefetch(Ahi + (size_t)(m0 + (tid >> 1)) * K + k0 + 32 + (tid & 1) * 16, 0, 3);
    async_wait_all();
    __syncthreads();

    // B fragments: lane<16 -> K {0..7,16..23} of col nl; lane>=16 -> K {8..15,24..31}
    v16bf bhi[2], blo[2];
    #pragma unroll
    for (int sn = 0; sn < 2; ++sn) {
      const __bf16* bh = &sWhi[wn * 32 + sn * 16 + nl][hi16 * 8];
      const __bf16* bl = &sWlo[wn * 32 + sn * 16 + nl][hi16 * 8];
      bhi[sn] = frag16(bh, bh + 16);
      blo[sn] = frag16(bl, bl + 16);
    }

    #pragma unroll
    for (int sm = 0; sm < 2; ++sm) {
      // A fragment: lane<16 -> K {0..7,16..23}; lane>=16 -> K {8..15,24..31}
      const __bf16* ah = &sAhi[wm * 32 + sm * 16 + nl][hi16 * 8];
      const __bf16* al = &sAlo[wm * 32 + sm * 16 + nl][hi16 * 8];
      v16bf ahi = frag16(ah, ah + 16);
      v16bf alo = frag16(al, al + 16);
      #pragma unroll
      for (int sn = 0; sn < 2; ++sn) {
        v8f a = acc[sm][sn];
        a = __builtin_amdgcn_wmma_f32_16x16x32_bf16(false, ahi, false, bhi[sn], (short)0, a, false, false);
        a = __builtin_amdgcn_wmma_f32_16x16x32_bf16(false, ahi, false, blo[sn], (short)0, a, false, false);
        a = __builtin_amdgcn_wmma_f32_16x16x32_bf16(false, alo, false, bhi[sn], (short)0, a, false, false);
        acc[sm][sn] = a;
      }
    }
    __syncthreads();
  }

  // epilogue: C layout — VGPR r holds M = r (lanes 0-15) or r+8 (lanes 16-31), N = lane&15
  #pragma unroll
  for (int sm = 0; sm < 2; ++sm) {
    #pragma unroll
    for (int sn = 0; sn < 2; ++sn) {
      const int gnc = n0 + wn * 32 + sn * 16 + nl;
      #pragma unroll
      for (int r = 0; r < 8; ++r) {
        int gm = m0 + wm * 32 + sm * 16 + hi16 * 8 + r;
        if (gm < M && gnc < N) {
          float v = acc[sm][sn][r];
          if (epi == 1) { v += bias[gnc]; v = (v > 20.f) ? v : log1pf(__expf(v)); }
          else if (epi == 2) { v += Radd[(size_t)gm * N + gnc]; }
          C[(size_t)gm * N + gnc] = v;
        }
      }
    }
  }
}

// ---------------- Causal depthwise conv1d (K=4) + SiLU -> bf16 pair ----------------
__global__ __launch_bounds__(256) void conv_silu_kernel(const float* __restrict__ x,
                                                        const float* __restrict__ conv_w,
                                                        const float* __restrict__ conv_b,
                                                        __bf16* __restrict__ xshi,
                                                        __bf16* __restrict__ xslo)
{
  const size_t idx = (size_t)blockIdx.x * blockDim.x + threadIdx.x;
  const int e = (int)(idx % E_);
  const int l = (int)((idx / E_) % L_);
  const int b = (int)(idx / ((size_t)E_ * L_));
  float acc = conv_b[e];
  #pragma unroll
  for (int k = 0; k < KC_; ++k) {
    int ls = l + k - (KC_ - 1);
    if (ls >= 0) acc += x[((size_t)(b * L_ + ls)) * E_ + e] * conv_w[e * KC_ + k];
  }
  float v = acc * sigmoidf_(acc);
  split_bf16(v, &xshi[idx], &xslo[idx]);
}

// ---------------- Fused selective-scan + gating epilogue ----------------
#define SCHUNK 64
__global__ __launch_bounds__(256) void ssm_scan_kernel(
    const float* __restrict__ delta,
    const __bf16* __restrict__ xshi, const __bf16* __restrict__ xslo,
    const float* __restrict__ Bt,    const float* __restrict__ Ct,
    const float* __restrict__ A_log, const float* __restrict__ W_D,
    const float* __restrict__ skip,
    __bf16* __restrict__ yphi, __bf16* __restrict__ yplo)
{
  const int tid = threadIdx.x;
  const int b   = blockIdx.x / (E_ / 256);
  const int e   = (blockIdx.x % (E_ / 256)) * 256 + tid;

  __shared__ float sB[SCHUNK][N_];
  __shared__ float sC[SCHUNK][N_];

  float An[N_], h[N_];
  #pragma unroll
  for (int n = 0; n < N_; ++n) { An[n] = -__expf(A_log[(size_t)e * N_ + n]); h[n] = 0.f; }
  const float wd = W_D[e];

  for (int lb = 0; lb < L_; lb += SCHUNK) {
    __syncthreads();
    for (int i = tid; i < SCHUNK * N_; i += 256) {
      size_t g = (size_t)(b * L_ + lb) * N_ + i;
      ((float*)sB)[i] = Bt[g];
      ((float*)sC)[i] = Ct[g];
    }
    __syncthreads();
    for (int li = 0; li < SCHUNK; ++li) {
      const size_t idx = ((size_t)(b * L_ + lb + li)) * E_ + e;
      const float dt = delta[idx];
      const float xt = (float)xshi[idx] + (float)xslo[idx];
      const float dx = dt * xt;
      float y = 0.f;
      #pragma unroll
      for (int n = 0; n < N_; ++n) {
        float ab = __expf(dt * An[n]);
        h[n] = ab * h[n] + dx * sB[li][n];
        y += h[n] * sC[li][n];
      }
      const float sk = skip[idx];
      float v = (y + xt * wd) * (sk * sigmoidf_(sk));
      split_bf16(v, &yphi[idx], &yplo[idx]);
    }
  }
}

// ---------------- Host-side launch ----------------
extern "C" void kernel_launch(void* const* d_in, const int* in_sizes, int n_in,
                              void* d_out, int out_size, void* d_ws, size_t ws_size,
                              hipStream_t stream)
{
  const float* resid  = (const float*)d_in[0];
  const float* norm_w = (const float*)d_in[1];
  const float* skip_w = (const float*)d_in[2];
  const float* in_w   = (const float*)d_in[3];
  const float* conv_w = (const float*)d_in[4];
  const float* conv_b = (const float*)d_in[5];
  const float* wd1    = (const float*)d_in[6];
  const float* wd2    = (const float*)d_in[7];
  const float* wd2_b  = (const float*)d_in[8];
  const float* wb     = (const float*)d_in[9];
  const float* wc     = (const float*)d_in[10];
  const float* A_log  = (const float*)d_in[11];
  const float* W_D    = (const float*)d_in[12];
  const float* out_w  = (const float*)d_in[13];
  float* out = (float*)d_out;

  // ---- workspace bump allocator (float units; bf16 arrays take n/2 floats) ----
  float* p = (float*)d_ws;
  auto allocF  = [&](size_t n) { float* q = p; p += n; return q; };
  auto allocB  = [&](size_t n) { __bf16* q = (__bf16*)p; p += (n + 1) / 2; return q; };

  float*  xproj = allocF((size_t)BL_ * E_);       // also reused as delta
  float*  skip  = allocF((size_t)BL_ * E_);
  float*  t1    = allocF((size_t)BL_ * DT_);
  float*  Btv   = allocF((size_t)BL_ * N_);
  float*  Ctv   = allocF((size_t)BL_ * N_);
  __bf16* xnhi  = allocB((size_t)BL_ * D_);
  __bf16* xnlo  = allocB((size_t)BL_ * D_);
  __bf16* iwh   = allocB((size_t)E_ * D_);
  __bf16* iwl   = allocB((size_t)E_ * D_);
  __bf16* swh   = allocB((size_t)E_ * D_);
  __bf16* swl   = allocB((size_t)E_ * D_);
  __bf16* owh   = allocB((size_t)D_ * E_);
  __bf16* owl   = allocB((size_t)D_ * E_);
  __bf16* w1h   = allocB((size_t)DT_ * E_);
  __bf16* w1l   = allocB((size_t)DT_ * E_);
  __bf16* w2h   = allocB((size_t)E_ * DT_);
  __bf16* w2l   = allocB((size_t)E_ * DT_);
  __bf16* wbh   = allocB((size_t)N_ * E_);
  __bf16* wbl   = allocB((size_t)N_ * E_);
  __bf16* wch   = allocB((size_t)N_ * E_);
  __bf16* wcl   = allocB((size_t)N_ * E_);
  __bf16* xsh   = allocB((size_t)BL_ * E_);
  __bf16* xsl   = allocB((size_t)BL_ * E_);
  __bf16* t1h   = allocB((size_t)BL_ * DT_);
  __bf16* t1l   = allocB((size_t)BL_ * DT_);
  __bf16* yph   = allocB((size_t)BL_ * E_);
  __bf16* ypl   = allocB((size_t)BL_ * E_);
  float*  delta = xproj;   // alias: xproj dead after conv

  dim3 blk(256);
  auto cvt = [&](const float* x, __bf16* h, __bf16* l, int n) {
    cvt_pair_kernel<<<(n + 255) / 256, blk, 0, stream>>>(x, h, l, n);
  };

  // weight decompositions (once per launch)
  cvt(in_w,   iwh, iwl, E_ * D_);
  cvt(skip_w, swh, swl, E_ * D_);
  cvt(out_w,  owh, owl, D_ * E_);
  cvt(wd1,    w1h, w1l, DT_ * E_);
  cvt(wd2,    w2h, w2l, E_ * DT_);
  cvt(wb,     wbh, wbl, N_ * E_);
  cvt(wc,     wch, wcl, N_ * E_);

  rmsnorm_kernel<<<BL_, blk, 0, stream>>>(resid, norm_w, xnhi, xnlo);

  gemm_bf16x3_kernel<<<dim3(BL_/128, E_/64), blk, 0, stream>>>(xnhi, xnlo, iwh, iwl, xproj, nullptr, nullptr, BL_, E_, D_, 0);
  gemm_bf16x3_kernel<<<dim3(BL_/128, E_/64), blk, 0, stream>>>(xnhi, xnlo, swh, swl, skip,  nullptr, nullptr, BL_, E_, D_, 0);

  conv_silu_kernel<<<(B_*L_*E_)/256, blk, 0, stream>>>(xproj, conv_w, conv_b, xsh, xsl);

  gemm_bf16x3_kernel<<<dim3(BL_/128, 1), blk, 0, stream>>>(xsh, xsl, w1h, w1l, t1, nullptr, nullptr, BL_, DT_, E_, 0);
  cvt(t1, t1h, t1l, BL_ * DT_);
  gemm_bf16x3_kernel<<<dim3(BL_/128, E_/64), blk, 0, stream>>>(t1h, t1l, w2h, w2l, delta, wd2_b, nullptr, BL_, E_, DT_, 1);
  gemm_bf16x3_kernel<<<dim3(BL_/128, 1), blk, 0, stream>>>(xsh, xsl, wbh, wbl, Btv, nullptr, nullptr, BL_, N_, E_, 0);
  gemm_bf16x3_kernel<<<dim3(BL_/128, 1), blk, 0, stream>>>(xsh, xsl, wch, wcl, Ctv, nullptr, nullptr, BL_, N_, E_, 0);

  ssm_scan_kernel<<<B_*(E_/256), blk, 0, stream>>>(delta, xsh, xsl, Btv, Ctv, A_log, W_D, skip, yph, ypl);

  gemm_bf16x3_kernel<<<dim3(BL_/128, D_/64), blk, 0, stream>>>(yph, ypl, owh, owl, out, nullptr, resid, BL_, D_, E_, 2);
}
